// WHVILinear_55070070669432
// MI455X (gfx1250) — matrix operand order repaired and checked by reading, not compiled, not run on hardware
//
#include <hip/hip_runtime.h>
#include <hip/hip_bf16.h>

// ---------------------------------------------------------------------------
// WHVI linear layer, restructured for MI455X (gfx1250):
//   u = mu + r @ (r^T @ eps)            (two matvecs, r = softplus(Rho))
//   v = fwht(u)                         (H is Sylvester-Hadamard: H=hd/32)
//   W[i,j] = s1[i]*s2[j]*v[i^j]/1024    (bf16, 2MB)
//   out    = relu(W @ x^T)              (bf16 WMMA GEMM, f32 accum,
//                                        TDM tensor_load_to_lds double buffer)
// ---------------------------------------------------------------------------

typedef __attribute__((ext_vector_type(16))) __bf16 v16bf;
typedef __attribute__((ext_vector_type(8)))  __bf16 v8bf;
typedef __attribute__((ext_vector_type(4)))  __bf16 v4bf;
typedef __attribute__((ext_vector_type(8)))  float  v8f;
typedef __attribute__((ext_vector_type(4)))  float  v4f;
typedef __attribute__((ext_vector_type(4)))  unsigned int u32x4;
typedef __attribute__((ext_vector_type(8)))  int          i32x8;
typedef __attribute__((ext_vector_type(4)))  int          i32x4;

#define DIM   1024
#define BATCH 8192

// CDNA5 Tensor Data Mover path (TENSORcnt-tracked). This toolchain declares
// the 6-argument form: (g0, g1, g2, g3, i32x8, cpol).
#if defined(__AMDGCN__) &&                              \
    __has_builtin(__builtin_amdgcn_tensor_load_to_lds) && \
    __has_builtin(__builtin_amdgcn_s_wait_tensorcnt)
#define USE_TDM 1
#else
#define USE_TDM 0
#endif

__device__ __forceinline__ float softplus_f(float x) {
    return (x > 20.0f) ? x : log1pf(expf(x));
}

// ---- x (f32, [B,D]) -> bf16, same layout -----------------------------------
__global__ void k_cvt_x(const float* __restrict__ x, __bf16* __restrict__ xbf) {
    int i = (blockIdx.x * 256 + threadIdx.x) * 8;
    v4f a = *(const v4f*)(x + i);
    v4f b = *(const v4f*)(x + i + 4);
    v8bf o;
#pragma unroll
    for (int q = 0; q < 4; ++q) { o[q] = (__bf16)a[q]; o[q + 4] = (__bf16)b[q]; }
    *(v8bf*)(xbf + i) = o;
}

// ---- t = softplus(Rho)^T @ eps  (column reduction, coalesced) --------------
__global__ void k_t(const float* __restrict__ Rho, const float* __restrict__ eps,
                    float* __restrict__ t) {
    int col = blockIdx.x * 256 + threadIdx.x;
    float acc = 0.0f;
    for (int j = 0; j < DIM; ++j)
        acc += softplus_f(Rho[j * DIM + col]) * eps[j];
    t[col] = acc;
}

// ---- u = mu + softplus(Rho) @ t  (one wave32 per row) ----------------------
__global__ void k_u(const float* __restrict__ Rho, const float* __restrict__ t,
                    const float* __restrict__ mu, float* __restrict__ u) {
    int row  = blockIdx.x * 8 + (threadIdx.x >> 5);
    int lane = threadIdx.x & 31;
    float acc = 0.0f;
    for (int j = lane; j < DIM; j += 32)
        acc += softplus_f(Rho[row * DIM + j]) * t[j];
#pragma unroll
    for (int off = 16; off > 0; off >>= 1)
        acc += __shfl_xor(acc, off, 32);
    if (lane == 0) u[row] = mu[row] + acc;
}

// ---- v = fwht(u)  (raw +-1 Walsh-Hadamard, single workgroup) ---------------
__global__ void k_fwht(const float* __restrict__ u, float* __restrict__ v) {
    __shared__ float s[DIM];
    int t = threadIdx.x;             // 1024 threads
    s[t] = u[t];
    __syncthreads();
    for (int bit = 1; bit < DIM; bit <<= 1) {
        float a = s[t];
        float b = s[t ^ bit];
        float n = (t & bit) ? (b - a) : (a + b);
        __syncthreads();
        s[t] = n;
        __syncthreads();
    }
    v[t] = s[t];
}

// ---- W_bf16[i,j] = s1[i]*s2[j]*v[i^j] / 1024 -------------------------------
__global__ void k_fillW(const float* __restrict__ s1, const float* __restrict__ s2,
                        const float* __restrict__ v, __bf16* __restrict__ W) {
    int idx = blockIdx.x * 256 + threadIdx.x;   // one thread per 4 elements
    int i   = idx >> 8;
    int jc  = (idx & 255) * 4;
    float si = s1[i] * (1.0f / 1024.0f);
    v4bf o;
#pragma unroll
    for (int q = 0; q < 4; ++q) {
        int j = jc + q;
        o[q] = (__bf16)(si * s2[j] * v[i ^ j]);
    }
    *(v4bf*)(W + i * DIM + jc) = o;
}

// ---- out = relu(W @ x^T) : bf16 WMMA, 128x128 WG tile, 8 waves (4x2) -------
// Double-buffered LDS; 128x32 bf16 tiles staged by one TDM descriptor each.
// LDS rows padded to 40 bf16 (80B): TDM pad = 4 DWORDs after every 16 DWORDs.
#define BK        32
#define LDS_PITCH 40

#if USE_TDM
// One tensor_load_to_lds: 2D tile [128 rows x 32 bf16] from a row-major
// [tensor_dim1 x 1024] bf16 tensor into LDS at lds_addr with 80B row pitch.
__device__ __forceinline__ void tdm_load_tile(const __bf16* gtile,
                                              unsigned lds_addr,
                                              unsigned tensor_rows) {
    unsigned long long ga = (unsigned long long)(uintptr_t)gtile;
    u32x4 g0;
    g0.x = 1u;                                   // count=1, user descriptor
    g0.y = lds_addr;                             // LDS byte address
    g0.z = (unsigned)ga;                         // global_addr[31:0]
    g0.w = (unsigned)((ga >> 32) & 0x01FFFFFFu)  // global_addr[56:32]
           | (2u << 30);                         // type = 2 ("image")
    i32x8 g1;
    g1[0] = 0x06D10000;          // data_size=2B, pad_enable, int=16DW, amt=4DW
    g1[1] = (int)(1024u << 16);  // tensor_dim0 = 1024  (bits[79:48] lo16)
    g1[2] = (int)(tensor_rows << 16); // tensor_dim1 lo16 (bits[111:80])
    g1[3] = (int)(32u << 16);    // tile_dim0 = 32 elements (64B rows)
    g1[4] = 128;                 // tile_dim1 = 128 rows, tile_dim2 = 0
    g1[5] = 1024;                // tensor_dim0_stride[31:0] = 1024 elements
    g1[6] = 0;                   // stride hi, dim1_stride lo (unused, 2D)
    g1[7] = 0;
    i32x4 z4 = {0, 0, 0, 0};     // groups 2/3 disabled (2D tensor)
    i32x8 z8 = {0, 0, 0, 0, 0, 0, 0, 0};
    __builtin_amdgcn_tensor_load_to_lds(g0, g1, z4, z4, z8, 0);
}
#endif

__global__ __launch_bounds__(256) void
k_gemm(const __bf16* __restrict__ W, const __bf16* __restrict__ X,
       float* __restrict__ out) {
    __shared__ __bf16 Ws[2][128 * LDS_PITCH];   // [m][k]  2 x 10KB
    __shared__ __bf16 Xs[2][128 * LDS_PITCH];   // [n][k]  2 x 10KB

    const int tid   = threadIdx.x;
    const int m0    = blockIdx.y * 128;      // D rows
    const int n0    = blockIdx.x * 128;      // batch cols
    const int wave  = tid >> 5;
    const int lane  = tid & 31;
    const int wm    = wave & 3;              // 4 wave-rows  (32 M each)
    const int wn    = wave >> 2;             // 2 wave-cols  (64 N each)
    const int h     = lane >> 4;             // K-half selector
    const int mr    = lane & 15;

    v8f acc[2][4];
#pragma unroll
    for (int i = 0; i < 2; ++i)
#pragma unroll
        for (int j = 0; j < 4; ++j) acc[i][j] = (v8f)(0.0f);

#if USE_TDM
    const unsigned ldsW[2] = {(unsigned)(uintptr_t)&Ws[0][0],
                              (unsigned)(uintptr_t)&Ws[1][0]};
    const unsigned ldsX[2] = {(unsigned)(uintptr_t)&Xs[0][0],
                              (unsigned)(uintptr_t)&Xs[1][0]};
    // prologue: stage K-tile 0 into buffer 0 (wave 0 drives the TDM)
    if (wave == 0) {
        tdm_load_tile(&W[(size_t)m0 * DIM], ldsW[0], 1024u);
        tdm_load_tile(&X[(size_t)n0 * DIM], ldsX[0], (unsigned)BATCH);
        __builtin_amdgcn_s_wait_tensorcnt(0);
    }
    __syncthreads();
#else
    // fallback: synchronous per-thread staging (2 b128 chunks per thread)
    const int r0 = tid >> 1;
    const int c0 = (tid & 1) * 8;
    const int c1 = c0 + 16;
    const __bf16* gW = &W[(m0 + r0) * DIM];
    const __bf16* gX = &X[(size_t)(n0 + r0) * DIM];
    *(v8bf*)&Ws[0][r0 * LDS_PITCH + c0] = *(const v8bf*)(gW + c0);
    *(v8bf*)&Ws[0][r0 * LDS_PITCH + c1] = *(const v8bf*)(gW + c1);
    *(v8bf*)&Xs[0][r0 * LDS_PITCH + c0] = *(const v8bf*)(gX + c0);
    *(v8bf*)&Xs[0][r0 * LDS_PITCH + c1] = *(const v8bf*)(gX + c1);
    __syncthreads();
#endif

    int cur = 0;
    for (int k0 = 0; k0 < DIM; k0 += BK) {
        // stage K-tile k0+BK into the other buffer while computing on `cur`
        if (k0 + BK < DIM) {
#if USE_TDM
            if (wave == 0) {
                tdm_load_tile(&W[(size_t)m0 * DIM + k0 + BK], ldsW[cur ^ 1], 1024u);
                tdm_load_tile(&X[(size_t)n0 * DIM + k0 + BK], ldsX[cur ^ 1],
                              (unsigned)BATCH);
            }
#else
            __bf16* dW = &Ws[cur ^ 1][r0 * LDS_PITCH];
            __bf16* dX = &Xs[cur ^ 1][r0 * LDS_PITCH];
            *(v8bf*)(dW + c0) = *(const v8bf*)(gW + k0 + BK + c0);
            *(v8bf*)(dW + c1) = *(const v8bf*)(gW + k0 + BK + c1);
            *(v8bf*)(dX + c0) = *(const v8bf*)(gX + k0 + BK + c0);
            *(v8bf*)(dX + c1) = *(const v8bf*)(gX + k0 + BK + c1);
#endif
        }

        const __bf16* bW = Ws[cur];
        const __bf16* bX = Xs[cur];

        // A fragments: lane holds row mr, K chunks {8h..8h+7} and {+16}
        v16bf afrag[2];
#pragma unroll
        for (int i = 0; i < 2; ++i) {
            const __bf16* p = &bW[(wm * 32 + i * 16 + mr) * LDS_PITCH + h * 8];
            v8bf lo = *(const v8bf*)(p);
            v8bf hi = *(const v8bf*)(p + 16);
            afrag[i] = __builtin_shufflevector(lo, hi, 0, 1, 2, 3, 4, 5, 6, 7,
                                               8, 9, 10, 11, 12, 13, 14, 15);
        }
        // B fragments: lane holds col mr, K chunks {8h..8h+7} and {+16}
        v16bf bfrag[4];
#pragma unroll
        for (int j = 0; j < 4; ++j) {
            const __bf16* p = &bX[(wn * 64 + j * 16 + mr) * LDS_PITCH + h * 8];
            v8bf lo = *(const v8bf*)(p);
            v8bf hi = *(const v8bf*)(p + 16);
            bfrag[j] = __builtin_shufflevector(lo, hi, 0, 1, 2, 3, 4, 5, 6, 7,
                                               8, 9, 10, 11, 12, 13, 14, 15);
        }
#pragma unroll
        for (int i = 0; i < 2; ++i)
#pragma unroll
            for (int j = 0; j < 4; ++j)
                acc[i][j] = __builtin_amdgcn_wmma_f32_16x16x32_bf16(
                    false, afrag[i], false, bfrag[j], (short)0, acc[i][j],
                    false, false);

#if USE_TDM
        if (wave == 0) __builtin_amdgcn_s_wait_tensorcnt(0);
#endif
        __syncthreads();     // DMA landed (wave0 waited), all reads done
        cur ^= 1;
    }

    // epilogue: C/D layout -> lane L: N = L%16, M = r + 8*(L/16)
    const int ncol = lane & 15;
    const int mofs = (lane >> 4) * 8;
#pragma unroll
    for (int i = 0; i < 2; ++i) {
#pragma unroll
        for (int j = 0; j < 4; ++j) {
            int col = n0 + wn * 64 + j * 16 + ncol;
#pragma unroll
            for (int r = 0; r < 8; ++r) {
                int row = m0 + wm * 32 + i * 16 + mofs + r;
                out[(size_t)row * BATCH + col] = fmaxf(acc[i][j][r], 0.0f);
            }
        }
    }
}

// ---------------------------------------------------------------------------
extern "C" void kernel_launch(void* const* d_in, const int* in_sizes, int n_in,
                              void* d_out, int out_size, void* d_ws, size_t ws_size,
                              hipStream_t stream) {
    (void)in_sizes; (void)n_in; (void)out_size; (void)ws_size;
    // setup_inputs order: x, s1, s2, mu, Rho, epsilon, H (H unused: structure exploited)
    const float* x   = (const float*)d_in[0];
    const float* s1  = (const float*)d_in[1];
    const float* s2  = (const float*)d_in[2];
    const float* mu  = (const float*)d_in[3];
    const float* Rho = (const float*)d_in[4];
    const float* eps = (const float*)d_in[5];
    float* out = (float*)d_out;

    char* ws = (char*)d_ws;
    __bf16* xbf = (__bf16*)ws;                                        // 16 MB
    __bf16* Wbf = (__bf16*)(ws + (size_t)BATCH * DIM * 2);            //  2 MB
    float*  t   = (float*)(ws + (size_t)BATCH * DIM * 2 + (size_t)DIM * DIM * 2);
    float*  u   = t + DIM;
    float*  v   = u + DIM;

    k_cvt_x<<<(BATCH * DIM) / (256 * 8), 256, 0, stream>>>(x, xbf);
    k_t    <<<DIM / 256, 256, 0, stream>>>(Rho, eps, t);
    k_u    <<<DIM / 8, 256, 0, stream>>>(Rho, t, mu, u);
    k_fwht <<<1, DIM, 0, stream>>>(u, v);
    k_fillW<<<(DIM * DIM / 4) / 256, 256, 0, stream>>>(s1, s2, v, Wbf);

    dim3 grid(BATCH / 128, DIM / 128);
    k_gemm <<<grid, 256, 0, stream>>>(Wbf, xbf, out);
}
